// DecoderSelfAttn_1425929142801
// MI455X (gfx1250) — matrix operand order repaired and checked by reading, not compiled, not run on hardware
//
#include <hip/hip_runtime.h>

// ---------------------------------------------------------------------------
// Fused decoder self-attention for MI455X (gfx1250, wave32, WMMA).
//   B=16 N=307 T=48 D=128 H=8 DK=16, conv kernel K=3, segments 12/12/24.
// One workgroup (8 waves) per (b,n); head == wave == GEMM N-tile everywhere.
// All GEMMs: v_wmma_f32_16x16x32_bf16. GEMMs iterate K outer / M inner with
// 3 accumulators so each weight fragment is loaded once. Row-softmax sums are
// computed as Pexp @ ones via WMMA (C-layout matches Pexp @ V exactly).
// Fragment masking redirects invalid lanes to a zeroed LDS block via integer
// offset selects -> no exec manipulation in the hot loops.
// ---------------------------------------------------------------------------

#define TT 48
#define DDIM 128

typedef __attribute__((ext_vector_type(16))) __bf16 bf16x16;
typedef __attribute__((ext_vector_type(8)))  float  f32x8;

union Frag {
  unsigned int   d[8];
  unsigned short u[16];
  bf16x16        v;
};

// LDS strides (ushort units): multiples of 8 (=> 16B-aligned rows) whose
// dword-stride spreads 16 consecutive rows over distinct banks.
#define SX_STR  136   // 48 x 136  staging tile / concat-heads buffer (272B rows)
#define SQK_STR 24    // per-head Q,K: 48 x 24  (48B rows)
#define SVT_STR 56    // per-head V^T: 16 x 56  (112B rows)
#define SP_STR  56    // per-head P:   48 x 56  (112B rows)

#define SZ_SX  (48 * SX_STR)          // 6528
#define OFF_SQ (SZ_SX)
#define SZ_QK  (8 * 48 * SQK_STR)     // 9216
#define OFF_SK (OFF_SQ + SZ_QK)
#define OFF_SV (OFF_SK + SZ_QK)
#define SZ_VT  (8 * 16 * SVT_STR)     // 7168
#define OFF_SP (OFF_SV + SZ_VT)
#define SZ_P   (8 * 48 * SP_STR)      // 21504
#define OFF_ZERO (OFF_SP + SZ_P)      // 32-ushort zero block (64B)
#define POOL_USHORTS (OFF_ZERO + 32)  // 53664 ushorts = 107328 B

// ws (bf16) regions: fragments of 32 lanes x 16 elems, B-matrix layout
#define NWQ (8*12*512)       // conv weights: 8 n-tiles x 12 k-blocks
#define NWV (8*4*512)        // linear weights: 8 n-tiles x 4 k-blocks
#define WS_TOTAL (2*NWQ + 2*NWV)   // 131072 bf16 = 256 KB

__device__ __forceinline__ unsigned short f2bf(float f) {
  union { __bf16 b; unsigned short u; } c;
  c.b = (__bf16)f;                       // RNE hardware convert
  return c.u;
}
__device__ __forceinline__ unsigned pack2bf(float a, float b) {
  union { __bf16 b2[2]; unsigned u; } c;
  c.b2[0] = (__bf16)a; c.b2[1] = (__bf16)b;
  return c.u;
}

__device__ __forceinline__ f32x8 zero8() {
  f32x8 z = {0.f,0.f,0.f,0.f,0.f,0.f,0.f,0.f};
  return z;
}

__device__ __forceinline__ f32x8 wmma_bf16(bf16x16 a, bf16x16 b, f32x8 c) {
  return __builtin_amdgcn_wmma_f32_16x16x32_bf16(false, a, false, b, (short)0, c,
                                                 false, false);
}

// reduce across the 16 lanes of each lane-half (xor bits 0..3 keep halves separate)
__device__ __forceinline__ float half16_max(float v) {
  int x;
  x = __builtin_amdgcn_ds_swizzle(__float_as_int(v), 0x041f); v = fmaxf(v, __int_as_float(x));
  x = __builtin_amdgcn_ds_swizzle(__float_as_int(v), 0x081f); v = fmaxf(v, __int_as_float(x));
  x = __builtin_amdgcn_ds_swizzle(__float_as_int(v), 0x101f); v = fmaxf(v, __int_as_float(x));
  x = __builtin_amdgcn_ds_swizzle(__float_as_int(v), 0x201f); v = fmaxf(v, __int_as_float(x));
  return v;
}

// ---- fragment loaders -----------------------------------------------------
// A layout: lane<16 row=lane, K {0..7,16..23}; lane>=16 row=lane-16, K {8..15,24..31}
// B layout: lane<16 col=lane, K {0..7,16..23}; lane>=16 col=lane-16, K {8..15,24..31}

// generic A fragment: run1 always valid; run2 (K +16..+23) optional at compile time
__device__ __forceinline__ bf16x16 load_a_frag(const unsigned short* mat, int stride,
                                               int m0, int kbase, int lane, bool run2) {
  const int row = m0 + (lane & 15);
  const int off = (lane >> 4) << 3;
  const unsigned short* p = mat + row * stride + kbase + off;
  Frag r;
  const uint4 lo = *(const uint4*)p;
  r.d[0] = lo.x; r.d[1] = lo.y; r.d[2] = lo.z; r.d[3] = lo.w;
  if (run2) {
    const uint4 hi = *(const uint4*)(p + 16);
    r.d[4] = hi.x; r.d[5] = hi.y; r.d[6] = hi.z; r.d[7] = hi.w;
  } else {
    r.d[4] = 0u; r.d[5] = 0u; r.d[6] = 0u; r.d[7] = 0u;
  }
  return r.v;
}

// im2col conv A: Xaug[t, kk=tap*128+din]; each 32-block sits in one tap.
// Invalid (pre-segment) lanes read the zero block: integer offset select only.
__device__ __forceinline__ bf16x16 load_a_conv(const unsigned short* pool,
                                               int m0, int kb, int lane) {
  const int row = m0 + (lane & 15);
  const int off = (lane >> 4) << 3;
  const int ts  = row - 2 + (kb >> 2);                      // causal source time
  const int seg = (row < 12) ? 0 : ((row < 24) ? 12 : 24);  // segment start
  const int dOff = ts * SX_STR + ((kb & 3) << 5) + off;     // computed always
  const int sel  = (ts >= seg) ? dOff : OFF_ZERO;           // one cndmask
  const unsigned short* p = pool + sel;
  Frag r;
  const uint4 lo = *(const uint4*)p;
  const uint4 hi = *(const uint4*)(p + 16);
  r.d[0] = lo.x; r.d[1] = lo.y; r.d[2] = lo.z; r.d[3] = lo.w;
  r.d[4] = hi.x; r.d[5] = hi.y; r.d[6] = hi.z; r.d[7] = hi.w;
  return r.v;
}

// packed weight fragment from global workspace (32B contiguous per lane)
__device__ __forceinline__ bf16x16 load_b_ws(const unsigned short* wsRegion,
                                             int frag, int lane) {
  return *(const bf16x16*)(wsRegion + (((frag << 5) + lane) << 4));
}

// 32B contiguous B fragment
__device__ __forceinline__ bf16x16 load_b16(const unsigned short* p) {
  Frag r;
  const uint4 a = *(const uint4*)p;
  const uint4 b = *(const uint4*)(p + 8);
  r.d[0] = a.x; r.d[1] = a.y; r.d[2] = a.z; r.d[3] = a.w;
  r.d[4] = b.x; r.d[5] = b.y; r.d[6] = b.z; r.d[7] = b.w;
  return r.v;
}

// scores B: B[k=dk, s] = K_h[s, dk]; DK=16 -> elements 8..15 (K 16..31) zero
__device__ __forceinline__ bf16x16 load_b_scoresK(const unsigned short* pool,
                                                  int khOff, int nt, int lane) {
  const unsigned short* p = pool + khOff + (nt * 16 + (lane & 15)) * SQK_STR + ((lane >> 4) << 3);
  Frag r;
  const uint4 lo = *(const uint4*)p;
  r.d[0] = lo.x; r.d[1] = lo.y; r.d[2] = lo.z; r.d[3] = lo.w;
  r.d[4] = 0u; r.d[5] = 0u; r.d[6] = 0u; r.d[7] = 0u;
  return r.v;
}

// P@V B: B[k=s, dk] = V^T_h[dk, s]; table K order along s. kb=1 elements 8..15 (s>=48) zero.
__device__ __forceinline__ bf16x16 load_b_attnv(const unsigned short* pool,
                                                int vtOff, int kb, int lane) {
  return load_a_frag(pool + vtOff, SVT_STR, 0, kb * 32, lane, kb == 0);
}

// constant ones B fragment for row-sum matmul (same zero pattern as V fragments)
__device__ __forceinline__ bf16x16 ones_b_frag(int kb, int lane) {
  const unsigned val = 0x3F803F80u;  // bf16 1.0 x2
  Frag r;
#pragma unroll
  for (int i = 0; i < 8; ++i) r.d[i] = (kb == 1 && i >= 4) ? 0u : val;
  return r.v;
}

// ---------------------------------------------------------------------------
// Prep: pack all four weight matrices into bf16 B-fragment order in d_ws.
//   conv:  Wc[k = tap*128+din, n = dout] = w[dout, din, tap]     (384x128)
//   lin :  Wl[k = d,            n = e   ] = w[e, d]              (128x128)
// ---------------------------------------------------------------------------
__global__ void prep_weights_kernel(const float* __restrict__ qw,
                                    const float* __restrict__ kw,
                                    const float* __restrict__ vw,
                                    const float* __restrict__ ow,
                                    unsigned short* __restrict__ ws) {
  int idx = blockIdx.x * blockDim.x + threadIdx.x;
  if (idx >= WS_TOTAL) return;
  int r = idx; const float* w; int isConv;
  if      (r < NWQ)            { w = qw; isConv = 1; }
  else if (r < 2 * NWQ)        { r -= NWQ;           w = kw; isConv = 1; }
  else if (r < 2 * NWQ + NWV)  { r -= 2 * NWQ;       w = vw; isConv = 0; }
  else                         { r -= 2 * NWQ + NWV; w = ow; isConv = 0; }
  const int f = r >> 9, rem = r & 511, l = rem >> 4, i = rem & 15;
  const int khalf = ((i >> 3) << 4) + ((l >> 4) << 3) + (i & 7) - i;
  float val;
  if (isConv) {
    const int nt = f / 12, kb = f % 12;
    const int k  = kb * 32 + khalf + i;       // 0..383
    const int n  = nt * 16 + (l & 15);        // dout
    const int tap = k >> 7, din = k & 127;
    val = w[(n * 128 + din) * 3 + tap];
  } else {
    const int nt = f >> 2, kb = f & 3;
    const int k  = kb * 32 + khalf + i;       // d
    const int e  = nt * 16 + (l & 15);
    val = w[e * 128 + k];
  }
  ws[idx] = f2bf(val);
}

// ---------------------------------------------------------------------------

__device__ __forceinline__ void fill_sX(unsigned short* sX,
                                        const float* __restrict__ src, int tid) {
  const float4* s4 = (const float4*)src;           // 48*128 f32 = 1536 float4
#pragma unroll
  for (int it = 0; it < 6; ++it) {
    const int i = tid + it * 256;
    const float4 q4 = s4[i];
    const int t = i >> 5;
    const int d = (i & 31) << 2;
    uint2 pk; pk.x = pack2bf(q4.x, q4.y); pk.y = pack2bf(q4.z, q4.w);
    *(uint2*)(sX + t * SX_STR + d) = pk;           // 8B-aligned packed store
  }
}

// causal-conv projection GEMM: 48x384 (im2col) @ 384x128, wave w -> head w.
// K outer / M inner: each weight fragment loaded once, 3 independent WMMAs.
__device__ __forceinline__ void conv_gemm(const unsigned short* pool,
                                          const unsigned short* wsW,
                                          const float* __restrict__ bias,
                                          unsigned short* dstHead, int lane, int w) {
  const int col = lane & 15;
  const int hi  = (lane >> 4) << 3;
  const float bc = bias[w * 16 + col];
  f32x8 acc0 = zero8(), acc1 = zero8(), acc2 = zero8();
#pragma unroll
  for (int kb = 0; kb < 12; ++kb) {
    const bf16x16 b = load_b_ws(wsW, w * 12 + kb, lane);
    acc0 = wmma_bf16(load_a_conv(pool,  0, kb, lane), b, acc0);
    acc1 = wmma_bf16(load_a_conv(pool, 16, kb, lane), b, acc1);
    acc2 = wmma_bf16(load_a_conv(pool, 32, kb, lane), b, acc2);
  }
#pragma unroll
  for (int rr = 0; rr < 8; ++rr) {
    const int t = rr + hi;
    dstHead[t * SQK_STR + col]        = f2bf(acc0[rr] + bc);
    dstHead[(t + 16) * SQK_STR + col] = f2bf(acc1[rr] + bc);
    dstHead[(t + 32) * SQK_STR + col] = f2bf(acc2[rr] + bc);
  }
}

// V projection GEMM: 48x128 @ 128x128; result stored TRANSPOSED (V^T[dk][s]),
// adjacent s pairs packed into b32 stores.
__device__ __forceinline__ void v_gemm(const unsigned short* sX,
                                       const unsigned short* wsW,
                                       const float* __restrict__ bias,
                                       unsigned short* sVTh, int lane, int w) {
  const int col = lane & 15;
  const int hi  = (lane >> 4) << 3;
  const float bc = bias[w * 16 + col];
  f32x8 acc0 = zero8(), acc1 = zero8(), acc2 = zero8();
#pragma unroll
  for (int kb = 0; kb < 4; ++kb) {
    const bf16x16 b = load_b_ws(wsW, w * 4 + kb, lane);
    acc0 = wmma_bf16(load_a_frag(sX, SX_STR,  0, kb * 32, lane, true), b, acc0);
    acc1 = wmma_bf16(load_a_frag(sX, SX_STR, 16, kb * 32, lane, true), b, acc1);
    acc2 = wmma_bf16(load_a_frag(sX, SX_STR, 32, kb * 32, lane, true), b, acc2);
  }
#pragma unroll
  for (int rr = 0; rr < 8; rr += 2) {             // s = t, t+1 adjacent in V^T row
    const int t = rr + hi;
    unsigned short* p = sVTh + col * SVT_STR;
    *(unsigned int*)(p + t)      = pack2bf(acc0[rr] + bc, acc0[rr + 1] + bc);
    *(unsigned int*)(p + t + 16) = pack2bf(acc1[rr] + bc, acc1[rr + 1] + bc);
    *(unsigned int*)(p + t + 32) = pack2bf(acc2[rr] + bc, acc2[rr + 1] + bc);
  }
}

__global__ __launch_bounds__(256) void fused_attn_kernel(
    const float* __restrict__ q, const float* __restrict__ k,
    const float* __restrict__ v,
    const float* __restrict__ qb, const float* __restrict__ kbias,
    const float* __restrict__ vb, const float* __restrict__ ob,
    const unsigned short* __restrict__ ws, float* __restrict__ out) {
  __shared__ __align__(16) unsigned short pool[POOL_USHORTS];

  const int tid  = threadIdx.x;
  const int lane = tid & 31;
  const int w    = tid >> 5;                  // wave id == head id == N-tile
  const int bn   = blockIdx.x;                // (b*N + n)
  const size_t base = (size_t)bn * TT * DDIM;

  unsigned short* sX = pool;                  // staging; reused as sO later
  unsigned short* sQ = pool + OFF_SQ;
  unsigned short* sK = pool + OFF_SK;
  unsigned short* sV = pool + OFF_SV;         // V^T
  unsigned short* sP = pool + OFF_SP;
  unsigned short* zrow = pool + OFF_ZERO;     // 64B of zeros

  const unsigned short* wsWq = ws;
  const unsigned short* wsWk = ws + NWQ;
  const unsigned short* wsWv = ws + 2 * NWQ;
  const unsigned short* wsWo = ws + 2 * NWQ + NWV;

  if (tid < 16) ((unsigned int*)zrow)[tid] = 0u;   // init zero block

  // ---- Q projection (causal conv) ----
  fill_sX(sX, q + base, tid);
  __syncthreads();
  conv_gemm(pool, wsWq, qb, sQ + w * 48 * SQK_STR, lane, w);
  __syncthreads();
  // ---- K projection (causal conv) ----
  fill_sX(sX, k + base, tid);
  __syncthreads();
  conv_gemm(pool, wsWk, kbias, sK + w * 48 * SQK_STR, lane, w);
  __syncthreads();
  // ---- V projection (linear, stored transposed) ----
  fill_sX(sX, v + base, tid);
  __syncthreads();
  v_gemm(sX, wsWv, vb, sV + w * 16 * SVT_STR, lane, w);
  __syncthreads();                            // sX now free -> alias as sO

  // ---- attention: entirely wave-private (head w) ----
  const int khOff = OFF_SK + w * 48 * SQK_STR;
  const int vtOff = OFF_SV + w * 16 * SVT_STR;
  const unsigned short* sQh = sQ + w * 48 * SQK_STR;
  unsigned short*       sPh = sP + w * 48 * SP_STR;
  const int col = lane & 15;
  const int hi8 = (lane >> 4) << 3;

#pragma unroll
  for (int mt = 0; mt < 3; ++mt) {
    // Q A-fragment: K-dim = 16 real + 16 zero pad (run2 = false)
    bf16x16 a = load_a_frag(sQh, SQK_STR, mt * 16, 0, lane, false);
    f32x8 sc0 = zero8(), sc1 = zero8(), sc2 = zero8();
    { bf16x16 b = load_b_scoresK(pool, khOff, 0, lane); sc0 = wmma_bf16(a, b, sc0); }
    if (mt >= 1) { bf16x16 b = load_b_scoresK(pool, khOff, 1, lane); sc1 = wmma_bf16(a, b, sc1); }
    if (mt >= 2) { bf16x16 b = load_b_scoresK(pool, khOff, 2, lane); sc2 = wmma_bf16(a, b, sc2); }
    // masked, max-subtracted exp; store UNNORMALIZED (sums come from WMMA below)
#pragma unroll
    for (int rr = 0; rr < 8; ++rr) {
      const int t = mt * 16 + rr + hi8;
      float v0 = sc0[rr] * 0.25f; if (col > t) v0 = -1e9f;
      float v1 = (mt >= 1) ? sc1[rr] * 0.25f : -1e9f;
      if (mt >= 1 && (16 + col) > t) v1 = -1e9f;
      float v2 = (mt >= 2) ? sc2[rr] * 0.25f : -1e9f;
      if (mt >= 2 && (32 + col) > t) v2 = -1e9f;
      const float rmax = half16_max(fmaxf(v0, fmaxf(v1, v2)));
      sPh[t * SP_STR + col]      = f2bf(__expf(v0 - rmax));
      sPh[t * SP_STR + 16 + col] = f2bf((mt >= 1) ? __expf(v1 - rmax) : 0.f);
      sPh[t * SP_STR + 32 + col] = f2bf((mt >= 2) ? __expf(v2 - rmax) : 0.f);
    }
  }

  // ---- Pexp @ V and Pexp @ ones (row sums) -> normalized heads into sO ----
  unsigned short* sO = sX;
  {
    f32x8 o0 = zero8(), o1 = zero8(), o2 = zero8();
    f32x8 s0 = zero8(), s1 = zero8(), s2 = zero8();
    {   // kb = 0 : s in 0..31
      const bf16x16 bv = load_b_attnv(pool, vtOff, 0, lane);
      const bf16x16 b1 = ones_b_frag(0, lane);
      const bf16x16 a0 = load_a_frag(sPh, SP_STR,  0, 0, lane, true);
      const bf16x16 a1 = load_a_frag(sPh, SP_STR, 16, 0, lane, true);
      const bf16x16 a2 = load_a_frag(sPh, SP_STR, 32, 0, lane, true);
      o0 = wmma_bf16(a0, bv, o0); s0 = wmma_bf16(a0, b1, s0);
      o1 = wmma_bf16(a1, bv, o1); s1 = wmma_bf16(a1, b1, s1);
      o2 = wmma_bf16(a2, bv, o2); s2 = wmma_bf16(a2, b1, s2);
    }
    {   // kb = 1 : s in 32..47 (+16 zero pad), only rows t>=32 have mass here
      const bf16x16 bv = load_b_attnv(pool, vtOff, 1, lane);
      const bf16x16 b1 = ones_b_frag(1, lane);
      const bf16x16 a2 = load_a_frag(sPh, SP_STR, 32, 32, lane, false);
      o2 = wmma_bf16(a2, bv, o2); s2 = wmma_bf16(a2, b1, s2);
    }
#pragma unroll
    for (int rr = 0; rr < 8; ++rr) {
      const int t = rr + hi8;
      sO[t * SX_STR + w * 16 + col]        = f2bf(o0[rr] / s0[rr]);
      sO[(t + 16) * SX_STR + w * 16 + col] = f2bf(o1[rr] / s1[rr]);
      sO[(t + 32) * SX_STR + w * 16 + col] = f2bf(o2[rr] / s2[rr]);
    }
  }
  __syncthreads();                            // all heads visible for O-proj

  // ---- output projection: 48x128 @ 128x128 + bias -> global f32 ----
  const float bo = ob[w * 16 + col];
  {
    f32x8 acc0 = zero8(), acc1 = zero8(), acc2 = zero8();
#pragma unroll
    for (int kb = 0; kb < 4; ++kb) {
      const bf16x16 b = load_b_ws(wsWo, w * 4 + kb, lane);
      acc0 = wmma_bf16(load_a_frag(sO, SX_STR,  0, kb * 32, lane, true), b, acc0);
      acc1 = wmma_bf16(load_a_frag(sO, SX_STR, 16, kb * 32, lane, true), b, acc1);
      acc2 = wmma_bf16(load_a_frag(sO, SX_STR, 32, kb * 32, lane, true), b, acc2);
    }
#pragma unroll
    for (int rr = 0; rr < 8; ++rr) {
      const int t = rr + hi8;
      float* o = out + base + w * 16 + col;
      o[(size_t)t * DDIM]        = acc0[rr] + bo;
      o[(size_t)(t + 16) * DDIM] = acc1[rr] + bo;
      o[(size_t)(t + 32) * DDIM] = acc2[rr] + bo;
    }
  }
}

// ---------------------------------------------------------------------------

extern "C" void kernel_launch(void* const* d_in, const int* in_sizes, int n_in,
                              void* d_out, int out_size, void* d_ws, size_t ws_size,
                              hipStream_t stream) {
  const float* q  = (const float*)d_in[0];
  const float* k  = (const float*)d_in[1];
  const float* v  = (const float*)d_in[2];
  // d_in[3] = mask (int tril) — causality implemented analytically
  const float* qw = (const float*)d_in[4];
  const float* qb = (const float*)d_in[5];
  const float* kw = (const float*)d_in[6];
  const float* kb = (const float*)d_in[7];
  const float* vw = (const float*)d_in[8];
  const float* vb = (const float*)d_in[9];
  const float* ow = (const float*)d_in[10];
  const float* ob = (const float*)d_in[11];
  unsigned short* ws = (unsigned short*)d_ws;
  float* out = (float*)d_out;

  prep_weights_kernel<<<(WS_TOTAL + 255) / 256, 256, 0, stream>>>(qw, kw, vw, ow, ws);
  fused_attn_kernel<<<16 * 307, 256, 0, stream>>>(q, k, v, qb, kb, vb, ob, ws, out);
}
